// HyperConvNet_19447611916692
// MI455X (gfx1250) — matrix operand order repaired and checked by reading, not compiled, not run on hardware
//
#include <hip/hip_runtime.h>
#include <math.h>

// ---------------------------------------------------------------------------
// MI455X (gfx1250) implementation. All large GEMMs go through
// v_wmma_f32_16x16x32_f16 (wave32). f32 operands are converted to f16 while
// staging tiles into LDS. A/B fragments use contiguous K order (k = 16*half + e).
// GEMM: 128xBN block tiles, 8 waves, float4 global staging, prefetch of the
// next K-slab (global_prefetch_b8). LSTM nonlinearities are branch-free
// (v_exp_f32 + v_rcp_f32) to keep the sequential recurrence divergence-free.
// ---------------------------------------------------------------------------

typedef __attribute__((ext_vector_type(16))) _Float16 v16h;
typedef __attribute__((ext_vector_type(8)))  float    v8f;

#define NNZ_E     262144
#define N_NODES_C 49152
#define B_G       4096
#define SEQ_T     34
#define PEP       320
#define XC_LD     4672   // fc input width: 34*128 + 320

__device__ __forceinline__ _Float16 f2h(float f) { return (_Float16)f; }

// branch-free sigmoid / tanh: v_exp_f32 + v_rcp_f32, no exec-mask divergence
__device__ __forceinline__ float sig_(float x) {
  return __builtin_amdgcn_rcpf(1.0f + __expf(-x));
}
__device__ __forceinline__ float tanh_(float x) {
  return 2.0f * __builtin_amdgcn_rcpf(1.0f + __expf(-2.0f * x)) - 1.0f;
}

__device__ __forceinline__ v8f wmma16(v16h a, v16h b, v8f c) {
  // (neg_a, A, neg_b, B, c_mod, C, reuse_a, reuse_b)
  return __builtin_amdgcn_wmma_f32_16x16x32_f16(false, a, false, b, (short)0, c,
                                                false, false);
}

// ---------------------------------------------------------------------------
// GEMM: C[M,N] = A[M,K](f32,row) * W[N,K](f32,row)^T + bias(+bias2), relu?
// grid = (M/128, N/BN), block = 256 threads (8 waves). K-step 32.
// Wave w: rows [ (w>>1)*32, +32 ), cols [ (w&1)*(BN/2), +(BN/2) ).
// ---------------------------------------------------------------------------
template <int BN>
__global__ void __launch_bounds__(256)
gemm_wmma_kernel(const float* __restrict__ A, int lda,
                 const float* __restrict__ W, int ldw,
                 float* __restrict__ C, int ldc,
                 const float* __restrict__ bias,
                 const float* __restrict__ bias2,
                 int K, int relu)
{
  constexpr int NT = BN / 32;        // 16-col tiles per wave
  const int m0   = blockIdx.x * 128;
  const int n0   = blockIdx.y * BN;
  const int tid  = threadIdx.x;
  const int wv   = tid >> 5;
  const int lane = tid & 31;
  const int half = lane >> 4;
  const int nl   = lane & 15;
  const int wm   = wv >> 1;          // 0..3 : 32-row slab
  const int wn   = wv & 1;           // 0..1 : BN/2-col slab

  __shared__ _Float16 As[128][36];   // [m][k], pitch 36 keeps f16x4 8B-aligned
  __shared__ _Float16 Bs[BN][36];    // [n][k]

  v8f acc[2][NT] = {};

  for (int k0 = 0; k0 < K; k0 += 32) {
    __syncthreads();
    // stage A tile: 128x32 f32 -> f16 (float4 loads)
    #pragma unroll
    for (int i = tid; i < 128 * 8; i += 256) {
      int r = i >> 3, c4 = (i & 7) << 2;
      const float4 v = *(const float4*)(A + (size_t)(m0 + r) * lda + k0 + c4);
      As[r][c4 + 0] = f2h(v.x); As[r][c4 + 1] = f2h(v.y);
      As[r][c4 + 2] = f2h(v.z); As[r][c4 + 3] = f2h(v.w);
    }
    // stage B tile: BNx32
    #pragma unroll
    for (int i = tid; i < BN * 8; i += 256) {
      int r = i >> 3, c4 = (i & 7) << 2;
      const float4 v = *(const float4*)(W + (size_t)(n0 + r) * ldw + k0 + c4);
      Bs[r][c4 + 0] = f2h(v.x); Bs[r][c4 + 1] = f2h(v.y);
      Bs[r][c4 + 2] = f2h(v.z); Bs[r][c4 + 3] = f2h(v.w);
    }
    // pull next K-slab toward the caches while this slab computes
    if (k0 + 32 < K) {
      int r = tid >> 3, c4 = (tid & 7) << 2;
      __builtin_prefetch(A + (size_t)(m0 + r) * lda + (k0 + 32) + c4, 0, 3);
      __builtin_prefetch(W + (size_t)(n0 + (r & (BN - 1))) * ldw + (k0 + 32) + c4, 0, 3);
    }
    __syncthreads();

    // A fragments (2 m-tiles). Contiguous K order matching B:
    // lanes 0-15: K=0..15 ; lanes 16-31: K=16..31
    v16h af[2];
    #pragma unroll
    for (int ii = 0; ii < 2; ii++)
      #pragma unroll
      for (int e = 0; e < 16; e++) {
        int ka = e + (half ? 16 : 0);
        af[ii][e] = As[wm * 32 + ii * 16 + nl][ka];
      }
    // B fragments (NT n-tiles). lanes 0-15: K=0..15 ; lanes 16-31: K=16..31
    v16h bfr[NT];
    #pragma unroll
    for (int j = 0; j < NT; j++)
      #pragma unroll
      for (int e = 0; e < 16; e++) {
        int kb = (half ? 16 + e : e);
        bfr[j][e] = Bs[wn * (BN / 2) + j * 16 + nl][kb];
      }
    #pragma unroll
    for (int i = 0; i < 2; i++)
      #pragma unroll
      for (int j = 0; j < NT; j++)
        acc[i][j] = wmma16(af[i], bfr[j], acc[i][j]);
  }

  #pragma unroll
  for (int i = 0; i < 2; i++) {
    #pragma unroll
    for (int j = 0; j < NT; j++) {
      int col = n0 + wn * (BN / 2) + j * 16 + nl;
      float bv = (bias ? bias[col] : 0.0f) + (bias2 ? bias2[col] : 0.0f);
      #pragma unroll
      for (int r = 0; r < 8; r++) {
        int row = m0 + wm * 32 + i * 16 + half * 8 + r;
        float v = acc[i][j][r] + bv;
        if (relu) v = fmaxf(v, 0.0f);
        C[(size_t)row * ldc + col] = v;
      }
    }
  }
}

// ---------------------------------------------------------------------------
// BiLSTM recurrence for one layer (both directions via blockIdx.y).
// G: precomputed gate inputs [B*T, 512], col = dir*256 + gate*64 + j
// Whh: [2,256,64]. Hout index = b*osB + t*osT + dir*64 + j.
// Block: 128 threads handles a 16-row batch strip; wave w owns h-cols
// [w*16, w*16+16) of each gate. c lives in registers, h in LDS (f16).
// ---------------------------------------------------------------------------
__global__ void __launch_bounds__(128)
lstm_rec_kernel(const float* __restrict__ G,
                const float* __restrict__ Whh,
                float* __restrict__ Hout, int osB, int osT)
{
  const int dir  = blockIdx.y;
  const int b0   = blockIdx.x * 16;
  const int tid  = threadIdx.x;
  const int wv   = tid >> 5;
  const int lane = tid & 31;
  const int half = lane >> 4;
  const int nl   = lane & 15;

  __shared__ _Float16 whh_s[256 * 64];  // [outcol][k]
  __shared__ _Float16 h_s[16 * 72];     // [m][k] padded

  const float* Wd = Whh + dir * 256 * 64;
  for (int i = tid; i < 256 * 64; i += 128) whh_s[i] = f2h(Wd[i]);
  for (int i = tid; i < 16 * 72; i += 128)  h_s[i]   = f2h(0.0f);
  __syncthreads();

  // Constant B fragments: gate q, K-chunk kc. B[k,n] = Whh[gcol=n, k].
  v16h bf[4][2];
  #pragma unroll
  for (int q = 0; q < 4; q++)
    #pragma unroll
    for (int kc = 0; kc < 2; kc++)
      #pragma unroll
      for (int e = 0; e < 16; e++) {
        int k = kc * 32 + (half ? 16 : 0) + e;
        bf[q][kc][e] = whh_s[(q * 64 + wv * 16 + nl) * 64 + k];
      }

  float c[8];
  #pragma unroll
  for (int r = 0; r < 8; r++) c[r] = 0.0f;

  for (int s = 0; s < SEQ_T; s++) {
    int t = dir ? (SEQ_T - 1 - s) : s;
    __syncthreads();  // h_s ready

    v16h af0, af1;
    #pragma unroll
    for (int e = 0; e < 16; e++) {
      int ka = e + (half ? 16 : 0);
      af0[e] = h_s[nl * 72 + ka];
      af1[e] = h_s[nl * 72 + 32 + ka];
    }

    v8f acc[4];
    #pragma unroll
    for (int q = 0; q < 4; q++) {
      acc[q] = v8f{};
      acc[q] = wmma16(af0, bf[q][0], acc[q]);
      acc[q] = wmma16(af1, bf[q][1], acc[q]);
    }

    float hv[8];
    #pragma unroll
    for (int r = 0; r < 8; r++) {
      int mr = half * 8 + r;
      int row = (b0 + mr) * SEQ_T + t;
      const float* gp = G + (size_t)row * 512 + dir * 256 + wv * 16 + nl;
      float gi = acc[0][r] + gp[0];
      float gf = acc[1][r] + gp[64];
      float gg = acc[2][r] + gp[128];
      float go = acc[3][r] + gp[192];
      float cn = sig_(gf) * c[r] + sig_(gi) * tanh_(gg);
      c[r] = cn;
      float h = sig_(go) * tanh_(cn);
      hv[r] = h;
      Hout[(size_t)(b0 + mr) * osB + (size_t)t * osT + dir * 64 + wv * 16 + nl] = h;
    }
    __syncthreads();  // all reads of h_s done
    #pragma unroll
    for (int r = 0; r < 8; r++)
      h_s[(half * 8 + r) * 72 + wv * 16 + nl] = f2h(hv[r]);
  }
}

// ---------------------------------------------------------------------------
// Hypergraph conv pieces
// ---------------------------------------------------------------------------
__global__ void deg_kernel(const int* __restrict__ node, const int* __restrict__ he,
                           float* __restrict__ deg_v, float* __restrict__ deg_e) {
  int i = blockIdx.x * blockDim.x + threadIdx.x;
  if (i < NNZ_E) {
    atomicAdd(&deg_v[node[i]], 1.0f);
    atomicAdd(&deg_e[he[i]], 1.0f);
  }
}

// xe[he] += xl[node]   (one wave per edge)
__global__ void __launch_bounds__(256)
scat_n2e_kernel(const float* __restrict__ xl, const int* __restrict__ node,
                const int* __restrict__ he, float* __restrict__ xe) {
  int e = blockIdx.x * 8 + (threadIdx.x >> 5);
  int lane = threadIdx.x & 31;
  if (e >= NNZ_E) return;
  const float* src = xl + (size_t)node[e] * PEP;
  float* dst = xe + (size_t)he[e] * PEP;
  for (int cc = lane; cc < PEP; cc += 32) atomicAdd(&dst[cc], src[cc]);
}

// xv[node] += Binv[he] * xe[he]
__global__ void __launch_bounds__(256)
scat_e2n_kernel(const float* __restrict__ xe, const float* __restrict__ deg_e,
                const int* __restrict__ node, const int* __restrict__ he,
                float* __restrict__ xv) {
  int e = blockIdx.x * 8 + (threadIdx.x >> 5);
  int lane = threadIdx.x & 31;
  if (e >= NNZ_E) return;
  int h = he[e];
  float d = deg_e[h];
  float binv = d > 0.0f ? 1.0f / d : 0.0f;
  const float* src = xe + (size_t)h * PEP;
  float* dst = xv + (size_t)node[e] * PEP;
  for (int cc = lane; cc < PEP; cc += 32) atomicAdd(&dst[cc], binv * src[cc]);
}

// in place: g = relu(Dinv * g + bias)
__global__ void fin_hconv_kernel(float* __restrict__ g, const float* __restrict__ deg_v,
                                 const float* __restrict__ bias) {
  int i = blockIdx.x * blockDim.x + threadIdx.x;
  if (i >= N_NODES_C * PEP) return;
  int r = i / PEP, cc = i % PEP;
  float d = deg_v[r];
  float dinv = d > 0.0f ? 1.0f / d : 0.0f;
  g[i] = fmaxf(dinv * g[i] + bias[cc], 0.0f);
}

// global max pool (12 consecutive nodes per graph) -> xc[:, 4352:4672]
__global__ void pool_kernel(const float* __restrict__ g, float* __restrict__ xc) {
  int i = blockIdx.x * blockDim.x + threadIdx.x;
  if (i >= B_G * PEP) return;
  int b = i / PEP, cc = i % PEP;
  float m = -INFINITY;
  #pragma unroll
  for (int k = 0; k < 12; k++)
    m = fmaxf(m, g[(size_t)(b * 12 + k) * PEP + cc]);
  xc[(size_t)b * XC_LD + (SEQ_T * 128) + cc] = m;
}

// final head: out[b,o] = h2[b,:] . outW[o,:] + outb[o]   (4096 x 2, K=128)
__global__ void head_kernel(const float* __restrict__ h2, const float* __restrict__ Wo,
                            const float* __restrict__ bo, float* __restrict__ out) {
  int i = blockIdx.x * blockDim.x + threadIdx.x;
  if (i >= B_G * 2) return;
  int row = i >> 1, o = i & 1;
  float s = bo[o];
  const float* hp = h2 + (size_t)row * 128;
  const float* wp = Wo + (size_t)o * 128;
  #pragma unroll 8
  for (int k = 0; k < 128; k++) s += hp[k] * wp[k];
  out[i] = s;
}

// ---------------------------------------------------------------------------
extern "C" void kernel_launch(void* const* d_in, const int* in_sizes, int n_in,
                              void* d_out, int out_size, void* d_ws, size_t ws_size,
                              hipStream_t stream) {
  (void)in_sizes; (void)n_in; (void)out_size; (void)ws_size;

  const float* x    = (const float*)d_in[0];
  const int*   eidx = (const int*)d_in[1];
  const float* mhc  = (const float*)d_in[2];
  // d_in[3] = batch (structure is b = node/12; not needed)
  const float* Wih0 = (const float*)d_in[4];
  const float* Whh0 = (const float*)d_in[5];
  const float* bih0 = (const float*)d_in[6];
  const float* bhh0 = (const float*)d_in[7];
  const float* Wih1 = (const float*)d_in[8];
  const float* Whh1 = (const float*)d_in[9];
  const float* bih1 = (const float*)d_in[10];
  const float* bhh1 = (const float*)d_in[11];
  const float* hgW1 = (const float*)d_in[12];
  const float* hgb1 = (const float*)d_in[13];
  const float* hgW2 = (const float*)d_in[14];
  const float* hgb2 = (const float*)d_in[15];
  const float* fc1W = (const float*)d_in[16];
  const float* fc1b = (const float*)d_in[17];
  const float* fc2W = (const float*)d_in[18];
  const float* fc2b = (const float*)d_in[19];
  const float* outW = (const float*)d_in[20];
  const float* outb = (const float*)d_in[21];
  float* out = (float*)d_out;

  const int* node = eidx;
  const int* he   = eidx + NNZ_E;

  // Workspace arena (f32 elements). hconv buffers reuse the retired G region.
  float* ws = (float*)d_ws;
  float* G  = ws;                 // 139264*512 = 71,303,168
  float* XL = ws;                 //  15,728,640 (alias, used after LSTM done)
  float* XE = ws + 15728640;      //  20,971,520
  float* GB = ws + 36700160;      //  15,728,640 (xv / g, in place)
  float* H0 = ws + 71303168;      //  17,825,792
  float* XC = ws + 89128960;      //  19,136,512
  float* F1 = ws + 108265472;     //   4,194,304
  float* F2 = ws + 112459776;     //     524,288
  float* DV = ws + 112984064;     //      49,152
  float* DE = ws + 113033216;     //      65,536  (total ~431.4 MiB)

  const int BT = B_G * SEQ_T;     // 139264

  // 1) G0 = mhc @ Wih0_cat^T + (bih0 + bhh0)   [BT,512], K=1280
  gemm_wmma_kernel<128><<<dim3(BT / 128, 512 / 128), 256, 0, stream>>>(
      mhc, 1280, Wih0, 1280, G, 512, bih0, bhh0, 1280, 0);

  // 2) layer-0 recurrence -> H0 [B,T,128]
  lstm_rec_kernel<<<dim3(B_G / 16, 2), 128, 0, stream>>>(G, Whh0, H0, SEQ_T * 128, 128);

  // 3) G1 = H0 @ Wih1_cat^T + (bih1 + bhh1)   [BT,512], K=128
  gemm_wmma_kernel<128><<<dim3(BT / 128, 512 / 128), 256, 0, stream>>>(
      H0, 128, Wih1, 128, G, 512, bih1, bhh1, 128, 0);

  // 4) layer-1 recurrence -> directly into XC cols [0, 4352)
  lstm_rec_kernel<<<dim3(B_G / 16, 2), 128, 0, stream>>>(G, Whh1, XC, XC_LD, 128);

  // 5) degrees (once)
  hipMemsetAsync(DV, 0, (size_t)(49152 + 65536) * 4, stream);
  deg_kernel<<<NNZ_E / 256, 256, 0, stream>>>(node, he, DV, DE);

  // 6) hypergraph conv layer 1
  gemm_wmma_kernel<64><<<dim3(N_NODES_C / 128, PEP / 64), 256, 0, stream>>>(
      x, PEP, hgW1, PEP, XL, PEP, nullptr, nullptr, PEP, 0);
  hipMemsetAsync(XE, 0, (size_t)65536 * PEP * 4, stream);
  scat_n2e_kernel<<<NNZ_E / 8, 256, 0, stream>>>(XL, node, he, XE);
  hipMemsetAsync(GB, 0, (size_t)N_NODES_C * PEP * 4, stream);
  scat_e2n_kernel<<<NNZ_E / 8, 256, 0, stream>>>(XE, DE, node, he, GB);
  fin_hconv_kernel<<<(N_NODES_C * PEP) / 256, 256, 0, stream>>>(GB, DV, hgb1);

  // 7) hypergraph conv layer 2 (GB -> XL -> XE -> GB in place)
  gemm_wmma_kernel<64><<<dim3(N_NODES_C / 128, PEP / 64), 256, 0, stream>>>(
      GB, PEP, hgW2, PEP, XL, PEP, nullptr, nullptr, PEP, 0);
  hipMemsetAsync(XE, 0, (size_t)65536 * PEP * 4, stream);
  scat_n2e_kernel<<<NNZ_E / 8, 256, 0, stream>>>(XL, node, he, XE);
  hipMemsetAsync(GB, 0, (size_t)N_NODES_C * PEP * 4, stream);
  scat_e2n_kernel<<<NNZ_E / 8, 256, 0, stream>>>(XE, DE, node, he, GB);
  fin_hconv_kernel<<<(N_NODES_C * PEP) / 256, 256, 0, stream>>>(GB, DV, hgb2);

  // 8) max pool -> XC cols [4352, 4672)
  pool_kernel<<<(B_G * PEP) / 256, 256, 0, stream>>>(GB, XC);

  // 9) FC1 (relu), FC2 (relu), head
  gemm_wmma_kernel<128><<<dim3(B_G / 128, 1024 / 128), 256, 0, stream>>>(
      XC, XC_LD, fc1W, XC_LD, F1, 1024, fc1b, nullptr, XC_LD, 1);
  gemm_wmma_kernel<128><<<dim3(B_G / 128, 128 / 128), 256, 0, stream>>>(
      F1, 1024, fc2W, 1024, F2, 128, fc2b, nullptr, 1024, 1);
  head_kernel<<<(B_G * 2) / 256, 256, 0, stream>>>(F2, outW, outb, out);
}
